// MoELayer_66116726555014
// MI455X (gfx1250) — matrix operand order repaired and checked by reading, not compiled, not run on hardware
//
#include <hip/hip_runtime.h>
#include <hip/hip_bf16.h>

typedef __attribute__((ext_vector_type(16))) __bf16 v16bf;
typedef __attribute__((ext_vector_type(8)))  float  v8f;

#define D_MODEL 1024
#define N_EXPERTS 8
#define HIDDEN 2048
#define NTOK 32768            // B*L = 8*4096
#define XSP 1032              // padded LDS row (bf16 elems): +16B/row -> bank rotation
#define BM 32                 // tokens per block (2 WMMA M-tiles)

// ---------------------------------------------------------------- init ------
__global__ void moe_init_kernel(int* counts) {
    if (threadIdx.x < N_EXPERTS) counts[threadIdx.x] = 0;
}

// ---------------------------------------------------------------- gate ------
__global__ void moe_gate_kernel(const float* __restrict__ x,
                                const float* __restrict__ gW,
                                const float* __restrict__ gb,
                                float* __restrict__ gate_out,   // [N, 8]
                                int*   __restrict__ counts,     // [E]
                                int*   __restrict__ toklist,    // [E, N]
                                float* __restrict__ topw)       // [N]
{
    const int lane = threadIdx.x & 31;
    const int t = blockIdx.x * (blockDim.x >> 5) + (threadIdx.x >> 5);
    if (t >= NTOK) return;

    float acc[N_EXPERTS];
#pragma unroll
    for (int e = 0; e < N_EXPERTS; ++e) acc[e] = 0.0f;

    const float* xr = x + (size_t)t * D_MODEL;
    for (int d = lane; d < D_MODEL; d += 32) {
        const float xv = xr[d];
        const float* wr = gW + (size_t)d * N_EXPERTS;
#pragma unroll
        for (int e = 0; e < N_EXPERTS; ++e) acc[e] += xv * wr[e];
    }
#pragma unroll
    for (int e = 0; e < N_EXPERTS; ++e) {
#pragma unroll
        for (int off = 16; off > 0; off >>= 1)
            acc[e] += __shfl_xor(acc[e], off, 32);
        acc[e] += gb[e];
    }
    float mx = acc[0];
#pragma unroll
    for (int e = 1; e < N_EXPERTS; ++e) mx = fmaxf(mx, acc[e]);
    float sum = 0.0f;
    float w[N_EXPERTS];
#pragma unroll
    for (int e = 0; e < N_EXPERTS; ++e) { w[e] = __expf(acc[e] - mx); sum += w[e]; }
    const float inv = 1.0f / sum;
    int best = 0; float bw = w[0];
#pragma unroll
    for (int e = 1; e < N_EXPERTS; ++e) { if (w[e] > bw) { bw = w[e]; best = e; } }

    if (lane < N_EXPERTS) gate_out[(size_t)t * N_EXPERTS + lane] = w[lane] * inv;
    if (lane == 0) {
        int pos = atomicAdd(&counts[best], 1);
        toklist[(size_t)best * NTOK + pos] = t;
        topw[t] = bw * inv;
    }
}

// --------------------------------------------------------- weight pack ------
// Repack W [E, K, Ncols] fp32 into WMMA-B-register-layout bf16 tiles:
// tile = (e, ct, kc) of shape 32(K) x 16(N); lane l owns column (l&15),
// K-run (l>>4)*16 + 0..15, stored as a contiguous 32B blob per lane.
__global__ void moe_pack_kernel(const float* __restrict__ W,
                                __bf16* __restrict__ Wp,
                                int K, int Ncols, int ntiles)
{
    const int tile = blockIdx.x * 8 + (threadIdx.x >> 5);
    if (tile >= ntiles) return;
    const int lane = threadIdx.x & 31;
    const int kt = K >> 5;
    const int kc = tile % kt;
    const int rest = tile / kt;
    const int ct = rest % (Ncols >> 4);
    const int e  = rest / (Ncols >> 4);

    const float* src = W + ((size_t)e * K + (size_t)kc * 32 + (lane >> 4) * 16) * Ncols
                         + ct * 16 + (lane & 15);
    __bf16* dst = Wp + (size_t)tile * 512 + lane * 16;
#pragma unroll
    for (int i = 0; i < 16; ++i) dst[i] = (__bf16)src[(size_t)i * Ncols];
}

// ----------------------------------------------------------------- ffn ------
// Block = 8 waves, 32 routed tokens (2 M-tiles) of one expert; each loaded
// B tile feeds 2 WMMAs (one per M-tile) to halve weight traffic per MAC.
template<bool PACKED>
__global__ void __launch_bounds__(256)
moe_ffn_kernel(const float* __restrict__ x,
               const float* __restrict__ W1, const float* __restrict__ b1,
               const float* __restrict__ W2, const float* __restrict__ b2,
               const __bf16* __restrict__ w1p, const __bf16* __restrict__ w2p,
               const int*   __restrict__ counts,
               const int*   __restrict__ toklist,
               const float* __restrict__ topw,
               float* __restrict__ out)
{
    const int e   = blockIdx.y;
    const int cnt = counts[e];
    const int m0  = blockIdx.x * BM;
    if (m0 >= cnt) return;

    __shared__ __bf16 xs[BM][XSP];   // 32 token rows of x (bf16)
    __shared__ __bf16 hs[BM][XSP];   // current 1024-wide chunk of h (bf16)
    __shared__ int   s_idx[BM];
    __shared__ float s_tw[BM];

    const int tid  = threadIdx.x;
    const int wave = tid >> 5;
    const int lane = tid & 31;
    const int ln16 = lane & 15;
    const int kh   = lane >> 4;

    if (tid < BM) {
        const int r = m0 + tid;
        if (r < cnt) {
            const int t = toklist[(size_t)e * NTOK + r];
            s_idx[tid] = t;
            s_tw[tid]  = topw[t];
        } else {
            s_idx[tid] = -1;
            s_tw[tid]  = 0.0f;
        }
    }
    __syncthreads();

    for (int i = tid; i < BM * D_MODEL; i += 256) {
        const int m = i >> 10, d = i & (D_MODEL - 1);
        const int t = s_idx[m];
        xs[m][d] = (__bf16)(t >= 0 ? x[(size_t)t * D_MODEL + d] : 0.0f);
    }
    __syncthreads();

    const float* W1e = W1 + (size_t)e * D_MODEL * HIDDEN;
    const float* W2e = W2 + (size_t)e * HIDDEN * D_MODEL;

    v8f accy[16] = {};   // y accumulators: 8 N-tiles x 2 M-tiles per wave

    for (int hc = 0; hc < 2; ++hc) {
        const int hbase = hc * 1024;

        // ---- GEMM1: h[:, hbase+0..1024) = relu(xs @ W1e) -------------------
        for (int ct = wave; ct < 64; ct += 8) {
            const int c0 = ct * 16;
            const int cg = (hbase + c0) >> 4;          // global column tile
            v8f acc0 = {}, acc1 = {};
            for (int k0 = 0; k0 < D_MODEL; k0 += 32) {
                v16bf b;
                if (PACKED) {
                    b = *(const v16bf*)(w1p
                        + ((((size_t)e * 128 + cg) * 32) + (k0 >> 5)) * 512
                        + lane * 16);
                } else {
                    const float* bp = W1e + (size_t)(k0 + kh * 16) * HIDDEN
                                          + hbase + c0 + ln16;
#pragma unroll
                    for (int i = 0; i < 16; ++i) b[i] = (__bf16)bp[(size_t)i * HIDDEN];
                }
                const __bf16* ap0 = &xs[ln16][k0 + kh * 16];
                const __bf16* ap1 = &xs[16 + ln16][k0 + kh * 16];
                v16bf a0, a1;
#pragma unroll
                for (int i = 0; i < 8; ++i) { a0[i] = ap0[i]; a0[8 + i] = ap0[8 + i]; }
#pragma unroll
                for (int i = 0; i < 8; ++i) { a1[i] = ap1[i]; a1[8 + i] = ap1[8 + i]; }
                acc0 = __builtin_amdgcn_wmma_f32_16x16x32_bf16(
                        false, a0, false, b, (short)0, acc0, false, false);
                acc1 = __builtin_amdgcn_wmma_f32_16x16x32_bf16(
                        false, a1, false, b, (short)0, acc1, false, false);
            }
            const float bb = b1[(size_t)e * HIDDEN + hbase + c0 + ln16];
#pragma unroll
            for (int r = 0; r < 8; ++r) {
                const int m = r + (kh << 3);
                const float v0 = acc0[r] + bb;
                const float v1 = acc1[r] + bb;
                hs[m][c0 + ln16]      = (__bf16)(v0 > 0.0f ? v0 : 0.0f);
                hs[16 + m][c0 + ln16] = (__bf16)(v1 > 0.0f ? v1 : 0.0f);
            }
        }
        __syncthreads();

        // ---- GEMM2 partial: y += hs @ W2e[hbase..] --------------------------
#pragma unroll
        for (int jt = 0; jt < 8; ++jt) {
            const int cg = wave * 8 + jt;              // column tile in D
            const int c0 = cg * 16;
            for (int k0 = 0; k0 < 1024; k0 += 32) {
                v16bf b;
                if (PACKED) {
                    const int kcg = (hbase + k0) >> 5;
                    b = *(const v16bf*)(w2p
                        + ((((size_t)e * 64 + cg) * 64) + kcg) * 512
                        + lane * 16);
                } else {
                    const float* bp = W2e + (size_t)(hbase + k0 + kh * 16) * D_MODEL
                                          + c0 + ln16;
#pragma unroll
                    for (int i = 0; i < 16; ++i) b[i] = (__bf16)bp[(size_t)i * D_MODEL];
                }
                const __bf16* ap0 = &hs[ln16][k0 + kh * 16];
                const __bf16* ap1 = &hs[16 + ln16][k0 + kh * 16];
                v16bf a0, a1;
#pragma unroll
                for (int i = 0; i < 8; ++i) { a0[i] = ap0[i]; a0[8 + i] = ap0[8 + i]; }
#pragma unroll
                for (int i = 0; i < 8; ++i) { a1[i] = ap1[i]; a1[8 + i] = ap1[8 + i]; }
                accy[2 * jt] = __builtin_amdgcn_wmma_f32_16x16x32_bf16(
                        false, a0, false, b, (short)0, accy[2 * jt], false, false);
                accy[2 * jt + 1] = __builtin_amdgcn_wmma_f32_16x16x32_bf16(
                        false, a1, false, b, (short)0, accy[2 * jt + 1], false, false);
            }
        }
        __syncthreads();
    }

    // ---- scale by top_w, add b2, scatter rows back ------------------------
#pragma unroll
    for (int jt = 0; jt < 8; ++jt) {
        const int c0 = (wave * 8 + jt) * 16;
        const float b2v = b2[(size_t)e * D_MODEL + c0 + ln16];
#pragma unroll
        for (int mt = 0; mt < 2; ++mt) {
#pragma unroll
            for (int r = 0; r < 8; ++r) {
                const int m = mt * 16 + r + (kh << 3);
                const int t = s_idx[m];
                if (t >= 0)
                    out[(size_t)t * D_MODEL + c0 + ln16] =
                        (accy[2 * jt + mt][r] + b2v) * s_tw[m];
            }
        }
    }
}

// -------------------------------------------------------------- launch -----
extern "C" void kernel_launch(void* const* d_in, const int* in_sizes, int n_in,
                              void* d_out, int out_size, void* d_ws, size_t ws_size,
                              hipStream_t stream) {
    const float* x  = (const float*)d_in[0];
    const float* gW = (const float*)d_in[1];
    const float* gb = (const float*)d_in[2];
    const float* W1 = (const float*)d_in[3];
    const float* b1 = (const float*)d_in[4];
    const float* W2 = (const float*)d_in[5];
    const float* b2 = (const float*)d_in[6];

    float* out      = (float*)d_out;                       // [8,4096,1024]
    float* gate_out = out + (size_t)NTOK * D_MODEL;        // [32768, 8]

    char*  ws      = (char*)d_ws;
    int*   counts  = (int*)ws;                              // 8 ints
    float* topw    = (float*)(ws + 1024);                   // N floats
    int*   toklist = (int*)(ws + 1024 + (size_t)NTOK * 4);  // E*N ints
    const size_t base = 1024 + (size_t)NTOK * 4 + (size_t)N_EXPERTS * NTOK * 4;

    const size_t wbytes = (size_t)N_EXPERTS * D_MODEL * HIDDEN * 2;  // 33.5 MB each
    const bool packed = ws_size >= base + 2 * wbytes;
    __bf16* w1p = (__bf16*)(ws + base);
    __bf16* w2p = (__bf16*)(ws + base + wbytes);

    moe_init_kernel<<<1, 32, 0, stream>>>(counts);
    moe_gate_kernel<<<NTOK / 8, 256, 0, stream>>>(x, gW, gb, gate_out,
                                                  counts, toklist, topw);

    dim3 grid(NTOK / BM, N_EXPERTS);
    if (packed) {
        const int nt1 = N_EXPERTS * (HIDDEN / 16) * (D_MODEL / 32);  // 32768
        const int nt2 = N_EXPERTS * (D_MODEL / 16) * (HIDDEN / 32);  // 32768
        moe_pack_kernel<<<(nt1 + 7) / 8, 256, 0, stream>>>(W1, w1p, D_MODEL, HIDDEN, nt1);
        moe_pack_kernel<<<(nt2 + 7) / 8, 256, 0, stream>>>(W2, w2p, HIDDEN, D_MODEL, nt2);
        moe_ffn_kernel<true><<<grid, 256, 0, stream>>>(x, W1, b1, W2, b2, w1p, w2p,
                                                       counts, toklist, topw, out);
    } else {
        moe_ffn_kernel<false><<<grid, 256, 0, stream>>>(x, W1, b1, W2, b2, w1p, w2p,
                                                        counts, toklist, topw, out);
    }
}
